// MPNN_1623497638117
// MI455X (gfx1250) — matrix operand order repaired and checked by reading, not compiled, not run on hardware
//
#include <hip/hip_runtime.h>
#include <cstdint>
#include <cstddef>

// ---------------- types ----------------
typedef __attribute__((ext_vector_type(16))) __bf16 v16bf;
typedef __attribute__((ext_vector_type(8)))  float  v8f;

#define N_IN   75
#define E_IN   6
#define H1     50
#define H2     80
#define DEG    32
#define KH     96      // global h row length (bf16), 3 WMMA K-steps
#define KHP    104     // LDS row stride for VwT (bank-skewed: 52 dw, gcd(52,64)=4)
#define KZ     160     // logical concat length, 5 WMMA K-steps
#define KZP    168     // LDS row stride for Z/UwT/RwT (84 dw, gcd(84,64)=4)
#define NWAVE  4
#define TPB    128

__device__ __forceinline__ __bf16 f2bf(float f) { return static_cast<__bf16>(f); }

__device__ __forceinline__ float selu_f(float x) {
  return 1.0507009873554805f * (x > 0.f ? x : 1.6732632423543772f * (__expf(x) - 1.f));
}

// A-fragment (16x32 bf16 tile): lane holds row m=lane&15; two 8-elem chunks at
// K = kb + k0 and kb + k0 + 16 with k0 = (lane>>4)*8  (per ISA 16-bit A layout)
__device__ __forceinline__ v16bf load_afrag(const __bf16* row, int kb, int k0) {
  union { v16bf v; uint4 q[2]; } u;
  u.q[0] = *(const uint4*)(row + kb + k0);
  u.q[1] = *(const uint4*)(row + kb + k0 + 16);
  return u.v;
}
// B-fragment (32x16 bf16 tile, stored as W^T rows): lane holds col n=lane&15;
// two 8-elem chunks at K = kb + (lane>>4)*8 and kb + (lane>>4)*8 + 16 (matches A)
__device__ __forceinline__ v16bf load_bfrag(const __bf16* row, int kb, int kh) {
  union { v16bf v; uint4 q[2]; } u;
  u.q[0] = *(const uint4*)(row + kb + (kh >> 1));
  u.q[1] = *(const uint4*)(row + kb + (kh >> 1) + 16);
  return u.v;
}

__device__ __forceinline__ v8f wmma_bf16(v16bf a, v16bf b, v8f c) {
  return __builtin_amdgcn_wmma_f32_16x16x32_bf16(false, a, false, b, (short)0, c, false, false);
}

__device__ __forceinline__ void wave_lds_sync() {
  asm volatile("s_wait_dscnt 0" ::: "memory");
}

// ---------------- prep kernels (bf16 conversion / transpose / pad) ----------------
__global__ void prep_h(const float* __restrict__ nf, __bf16* __restrict__ hA,
                       __bf16* __restrict__ hB, float* __restrict__ gacc, int n_nodes) {
  size_t i = (size_t)blockIdx.x * blockDim.x + threadIdx.x;
  size_t tot = (size_t)n_nodes * KH;
  if (i < tot) {
    int k = (int)(i % KH);
    size_t node = i / KH;
    hA[i] = (k < N_IN) ? f2bf(nf[node * N_IN + k]) : f2bf(0.f);
    hB[i] = f2bf(0.f);
  }
  if (i < H2) gacc[i] = 0.f;
}

__global__ void prep_V(const float* __restrict__ Vw, __bf16* __restrict__ out) {
  int i = blockIdx.x * blockDim.x + threadIdx.x;
  if (i >= 3 * 64 * KHP) return;
  int l = i / (64 * KHP); int rem = i - l * 64 * KHP;
  int n = rem / KHP; int k = rem - n * KHP;
  float v = (n < H1 && k < N_IN) ? Vw[(size_t)l * N_IN * H1 + (size_t)k * H1 + n] : 0.f;
  out[i] = f2bf(v);
}

__global__ void prep_U(const float* __restrict__ Uw, __bf16* __restrict__ out) {
  int i = blockIdx.x * blockDim.x + threadIdx.x;
  if (i >= 3 * 80 * KZP) return;
  int l = i / (80 * KZP); int rem = i - l * 80 * KZP;
  int n = rem / KZP; int p = rem - n * KZP;
  int k = -1;
  if (p < N_IN) k = p;                       // h part  (0..74)
  else if (p >= 80 && p < 136) k = p - 5;    // m_w(75..124) + m_e(125..130)
  float v = 0.f;
  if (n < N_IN && k >= 0) v = Uw[(size_t)l * 131 * N_IN + (size_t)k * N_IN + n];
  out[i] = f2bf(v);
}

__global__ void prep_R(const float* __restrict__ Rw, __bf16* __restrict__ out) {
  int i = blockIdx.x * blockDim.x + threadIdx.x;
  if (i >= 80 * KZP) return;
  int n = i / KZP; int p = i - n * KZP;
  int k = -1;
  if (p < N_IN) k = p;                       // h part
  else if (p >= 80 && p < 155) k = p - 5;    // x0 part (75..149)
  float v = (k >= 0) ? Rw[(size_t)k * H2 + n] : 0.f;
  out[i] = f2bf(v);
}

// ---------------- per-slot message passing step ----------------
__global__ __launch_bounds__(TPB) void slot_step(
    const __bf16* __restrict__ hsrc, __bf16* __restrict__ hdst,
    const int* __restrict__ neighbors, const float* __restrict__ edge_feat,
    const __bf16* __restrict__ VwT, const float* __restrict__ Vb,
    const __bf16* __restrict__ UwT, const float* __restrict__ Ub,
    const float* __restrict__ Ew, const float* __restrict__ Eb,
    int slot, int ntiles)
{
  __shared__ __bf16 sV[64 * KHP];          // VwT padded: 64 x 104
  __shared__ __bf16 sU[80 * KZP];          // UwT padded: 80 x 168
  __shared__ __bf16 sZ[NWAVE][16 * KZP];   // concat tiles (also output staging)

  {
    const unsigned* gV = (const unsigned*)VwT; unsigned* lV = (unsigned*)sV;
    for (int i = threadIdx.x; i < 64 * KHP / 2; i += TPB) lV[i] = gV[i];
    const unsigned* gU = (const unsigned*)UwT; unsigned* lU = (unsigned*)sU;
    for (int i = threadIdx.x; i < 80 * KZP / 2; i += TPB) lU[i] = gU[i];
  }
  __syncthreads();

  const int w    = threadIdx.x >> 5;
  const int lane = threadIdx.x & 31;
  const int tile = blockIdx.x * NWAVE + w;
  if (tile >= ntiles) return;
  const int base = tile * 16;
  const int mrow = lane & 15;
  const int half = lane >> 4;
  const int k0   = half * 8;    // A chunk offset
  const int kh   = half * 16;   // B chunk offset

  __bf16* Z = &sZ[w][0];

  // own h rows -> Z[:, 0:80)   (pad cols 75..79 are already zero in hsrc rows)
  {
    const uint4* s4 = (const uint4*)hsrc;   // h row = 12 uint4
    uint4* z4 = (uint4*)Z;                  // Z row = 21 uint4
    #pragma unroll
    for (int t = 0; t < 5; ++t) {
      int idx = t * 32 + lane;              // 0..159 = 16 rows x 10 chunks
      int r = idx / 10, c = idx - r * 10;
      z4[r * 21 + c] = s4[(size_t)(base + r) * 12 + c];
    }
  }

  // edge MLP (6->6) -> Z[:, 130:136)
  if (lane < 16) {
    const float* ef = edge_feat + ((size_t)(base + lane) * DEG + slot) * E_IN;
    float e0 = ef[0], e1 = ef[1], e2 = ef[2], e3 = ef[3], e4 = ef[4], e5 = ef[5];
    #pragma unroll
    for (int j = 0; j < E_IN; ++j) {
      float o = Eb[j] + e0 * Ew[0 * 6 + j] + e1 * Ew[1 * 6 + j] + e2 * Ew[2 * 6 + j]
                      + e3 * Ew[3 * 6 + j] + e4 * Ew[4 * 6 + j] + e5 * Ew[5 * 6 + j];
      Z[lane * KZP + 130 + j] = f2bf(o);
    }
  }

  // gather neighbor h row (L2-resident) directly into A fragments
  const int nb = neighbors[(size_t)(base + mrow) * DEG + slot];
  const __bf16* arow = hsrc + (size_t)nb * KH;
  v16bf ag[3];
  #pragma unroll
  for (int ks = 0; ks < 3; ++ks) ag[ks] = load_afrag(arow, ks * 32, k0);

  // m_w = h[nb] @ Vw + Vb   (4 n-tiles x 3 k-steps) -> Z[:, 80:130)
  #pragma unroll
  for (int tn = 0; tn < 4; ++tn) {
    const int col = tn * 16 + mrow;
    const float bias = (col < H1) ? Vb[col] : 0.f;
    v8f c;
    #pragma unroll
    for (int i = 0; i < 8; ++i) c[i] = bias;
    const __bf16* brow = &sV[col * KHP];
    #pragma unroll
    for (int ks = 0; ks < 3; ++ks)
      c = wmma_bf16(ag[ks], load_bfrag(brow, ks * 32, kh), c);
    if (col < H1) {
      #pragma unroll
      for (int i = 0; i < 8; ++i)
        Z[(i + 8 * half) * KZP + 80 + col] = f2bf(c[i]);
    }
  }

  wave_lds_sync();

  // z = concat @ Uw + Ub ; h' = selu(z)   (5 n-tiles x 5 k-steps)
  v16bf az[5];
  const __bf16* zrow = Z + mrow * KZP;
  #pragma unroll
  for (int ks = 0; ks < 5; ++ks) az[ks] = load_afrag(zrow, ks * 32, k0);

  wave_lds_sync();   // az captured; Z is now dead -> reuse as output staging

  // zero pad cols [80, 96) of the staging region (cols 75..79 already zero)
  {
    unsigned* zd = (unsigned*)Z;           // Z row = 84 dwords; col80 -> dword 40
    #pragma unroll
    for (int t = 0; t < 4; ++t) {          // 16 rows x 8 dwords = 128
      int idx = t * 32 + lane;
      int r = idx >> 3, c = idx & 7;
      zd[r * 84 + 40 + c] = 0u;
    }
  }

  #pragma unroll
  for (int tn = 0; tn < 5; ++tn) {
    const int col = tn * 16 + mrow;
    const float bias = (col < N_IN) ? Ub[col] : 0.f;
    v8f c;
    #pragma unroll
    for (int i = 0; i < 8; ++i) c[i] = bias;
    const __bf16* brow = &sU[col * KZP];
    #pragma unroll
    for (int ks = 0; ks < 5; ++ks)
      c = wmma_bf16(az[ks], load_bfrag(brow, ks * 32, kh), c);
    if (col < N_IN) {
      #pragma unroll
      for (int i = 0; i < 8; ++i)
        Z[(i + 8 * half) * KZP + col] = f2bf(selu_f(c[i]));   // stage h' in LDS
    }
  }

  wave_lds_sync();

  // coalesced write-back: 16 rows x 96 bf16 = 12 uint4 per row
  {
    const uint4* z4 = (const uint4*)Z;     // Z row = 21 uint4
    uint4* d4 = (uint4*)hdst;              // h row = 12 uint4
    #pragma unroll
    for (int t = 0; t < 6; ++t) {          // 192 chunks / 32 lanes
      int idx = t * 32 + lane;
      int r = idx / 12, c = idx - r * 12;
      d4[(size_t)(base + r) * 12 + c] = z4[r * 21 + c];
    }
  }
}

// ---------------- readout: selu([h,x0] @ R + Rb) summed over nodes ----------------
__global__ __launch_bounds__(TPB) void readout_step(
    const __bf16* __restrict__ h, const float* __restrict__ nf,
    const __bf16* __restrict__ RwT, const float* __restrict__ Rb,
    float* __restrict__ gacc, int ntiles)
{
  __shared__ __bf16 sR[80 * KZP];
  __shared__ __bf16 sZ[NWAVE][16 * KZP];
  __shared__ float sAcc[H2];

  if (threadIdx.x < H2) sAcc[threadIdx.x] = 0.f;
  {
    const unsigned* gR = (const unsigned*)RwT; unsigned* lR = (unsigned*)sR;
    for (int i = threadIdx.x; i < 80 * KZP / 2; i += TPB) lR[i] = gR[i];
  }
  __syncthreads();

  const int w = threadIdx.x >> 5, lane = threadIdx.x & 31;
  const int tile = blockIdx.x * NWAVE + w;
  if (tile < ntiles) {
    const int base = tile * 16;
    const int mrow = lane & 15;
    const int k0 = (lane >> 4) * 8, kh = (lane >> 4) * 16;
    __bf16* Z = &sZ[w][0];

    const uint4* h4 = (const uint4*)h;
    uint4* z4 = (uint4*)Z;
    #pragma unroll
    for (int t = 0; t < 5; ++t) {
      int idx = t * 32 + lane; int r = idx / 10, c = idx - r * 10;
      z4[r * 21 + c] = h4[(size_t)(base + r) * 12 + c];
    }
    for (int t = lane; t < 16 * 80; t += 32) {  // x0 -> Z[:, 80:160)
      int r = t / 80, c = t - r * 80;
      Z[r * KZP + 80 + c] = (c < N_IN) ? f2bf(nf[(size_t)(base + r) * N_IN + c]) : f2bf(0.f);
    }
    wave_lds_sync();

    v16bf az[5];
    const __bf16* zrow = Z + mrow * KZP;
    #pragma unroll
    for (int ks = 0; ks < 5; ++ks) az[ks] = load_afrag(zrow, ks * 32, k0);

    #pragma unroll
    for (int tn = 0; tn < 5; ++tn) {
      const int col = tn * 16 + mrow;
      const float bias = Rb[col];
      v8f c;
      #pragma unroll
      for (int i = 0; i < 8; ++i) c[i] = bias;
      const __bf16* brow = &sR[col * KZP];
      #pragma unroll
      for (int ks = 0; ks < 5; ++ks)
        c = wmma_bf16(az[ks], load_bfrag(brow, ks * 32, kh), c);
      float s = 0.f;
      #pragma unroll
      for (int i = 0; i < 8; ++i) s += selu_f(c[i]);
      atomicAdd(&sAcc[col], s);
    }
  }
  __syncthreads();
  if (threadIdx.x < H2) atomicAdd(&gacc[threadIdx.x], sAcc[threadIdx.x]);
}

__global__ void finalize_k(const float* __restrict__ gacc, const float* __restrict__ l0w,
                           const float* __restrict__ l0b, float* __restrict__ out) {
  __shared__ float red[128];
  float v = 0.f;
  if (threadIdx.x < H2) v = tanhf(gacc[threadIdx.x]) * l0w[threadIdx.x];
  red[threadIdx.x] = v;
  __syncthreads();
  for (int s = 64; s > 0; s >>= 1) {
    if (threadIdx.x < s) red[threadIdx.x] += red[threadIdx.x + s];
    __syncthreads();
  }
  if (threadIdx.x == 0) out[0] = red[0] + l0b[0];
}

// ---------------- host launcher ----------------
extern "C" void kernel_launch(void* const* d_in, const int* in_sizes, int n_in,
                              void* d_out, int out_size, void* d_ws, size_t ws_size,
                              hipStream_t stream) {
  const float* nf  = (const float*)d_in[0];
  const int*   nbr = (const int*)d_in[1];
  const float* ef  = (const float*)d_in[2];
  const float* Vw  = (const float*)d_in[3];
  const float* Vb  = (const float*)d_in[4];
  const float* Uw  = (const float*)d_in[5];
  const float* Ub  = (const float*)d_in[6];
  const float* Ew  = (const float*)d_in[7];
  const float* Eb  = (const float*)d_in[8];
  const float* Rw  = (const float*)d_in[9];
  const float* Rb  = (const float*)d_in[10];
  const float* l0w = (const float*)d_in[11];
  const float* l0b = (const float*)d_in[12];
  float* out = (float*)d_out;

  const int n_nodes = in_sizes[0] / N_IN;   // 200000 (multiple of 16)
  const int ntiles  = n_nodes / 16;

  char* ws = (char*)d_ws;
  size_t off = 0;
  auto alloc = [&](size_t bytes) {
    void* p = ws + off;
    off = (off + bytes + 255) & ~(size_t)255;
    return p;
  };
  __bf16* hA  = (__bf16*)alloc((size_t)n_nodes * KH * 2);
  __bf16* hB  = (__bf16*)alloc((size_t)n_nodes * KH * 2);
  __bf16* wV  = (__bf16*)alloc((size_t)3 * 64 * KHP * 2);
  __bf16* wU  = (__bf16*)alloc((size_t)3 * 80 * KZP * 2);
  __bf16* wR  = (__bf16*)alloc((size_t)80 * KZP * 2);
  float*  acc = (float*)alloc(H2 * sizeof(float));

  // weight / state prep (re-run every call -> deterministic)
  {
    size_t tot = (size_t)n_nodes * KH;
    prep_h<<<(unsigned)((tot + 255) / 256), 256, 0, stream>>>(nf, hA, hB, acc, n_nodes);
    prep_V<<<(3 * 64 * KHP + 255) / 256, 256, 0, stream>>>(Vw, wV);
    prep_U<<<(3 * 80 * KZP + 255) / 256, 256, 0, stream>>>(Uw, wU);
    prep_R<<<(80 * KZP + 255) / 256, 256, 0, stream>>>(Rw, wR);
  }

  const int grid = (ntiles + NWAVE - 1) / NWAVE;
  __bf16* src = hA;
  __bf16* dst = hB;
  for (int l = 0; l < 3; ++l) {
    const __bf16* VwT = wV + (size_t)l * 64 * KHP;
    const __bf16* UwT = wU + (size_t)l * 80 * KZP;
    const float* Vbl = Vb + l * H1;
    const float* Ubl = Ub + l * N_IN;
    for (int s = 0; s < DEG; ++s) {
      slot_step<<<grid, TPB, 0, stream>>>(src, dst, nbr, ef, VwT, Vbl, UwT, Ubl, Ew, Eb, s, ntiles);
      __bf16* t = src; src = dst; dst = t;
    }
  }
  readout_step<<<grid, TPB, 0, stream>>>(src, nf, wR, Rb, acc, ntiles);
  finalize_k<<<1, 128, 0, stream>>>(acc, l0w, l0b, out);
}
